// Expert_22634477650072
// MI455X (gfx1250) — matrix operand order repaired and checked by reading, not compiled, not run on hardware
//
#include <hip/hip_runtime.h>
#include <math.h>

// ---------------- problem constants ----------------
#define NE 8          // experts
#define TOPK 2
#define T_TOK 4096    // BATCH*SEQ
#define DM 1024       // d_model
#define DH 4096       // d_hid
#define PAIRS (T_TOK * TOPK)        // 8192
#define MTILE 32                    // pair rows per block (expert lists padded to 32)
#define MAXP (PAIRS + MTILE * NE)   // 8448 padded pair slots upper bound
#define MAXTILES (MAXP / MTILE)     // 264

// ---------------- types ----------------
typedef __bf16 bf16_t;
typedef bf16_t v16bf __attribute__((ext_vector_type(16)));
typedef bf16_t v8bf  __attribute__((ext_vector_type(8)));
typedef float  v8f   __attribute__((ext_vector_type(8)));

static __device__ __forceinline__ unsigned short f2bf(float f) {
    // round-to-nearest-even fp32 -> bf16
    unsigned u = __float_as_uint(f);
    unsigned r = u + 0x7FFFu + ((u >> 16) & 1u);
    return (unsigned short)(r >> 16);
}

static __device__ __forceinline__ v16bf cat16(v8bf lo, v8bf hi) {
    return __builtin_shufflevector(lo, hi, 0, 1, 2, 3, 4, 5, 6, 7,
                                           8, 9, 10, 11, 12, 13, 14, 15);
}
static __device__ __forceinline__ v8bf ldb8(const unsigned short* p) {
    return *(const v8bf*)p;
}

// ---------------- kernel: zero output ----------------
__global__ void k_zero(float* __restrict__ out, int n) {
    int i = blockIdx.x * blockDim.x + threadIdx.x;
    if (i < n) out[i] = 0.f;
}

// ---------------- kernel: x (fp32) -> xb (bf16, row-major) ----------------
__global__ void k_convx(const float* __restrict__ x, unsigned short* __restrict__ xb, int n) {
    int i = (blockIdx.x * blockDim.x + threadIdx.x) * 4;
    if (i < n) {
#pragma unroll
        for (int j = 0; j < 4; ++j) xb[i + j] = f2bf(x[i + j]);
    }
}

// ---------------- kernel: weights fp32 [E][K][N] -> bf16 swizzled B-fragment tiles --
// dst tile (e, nt, kt): 512 bf16, element [l*16 + i] = src[e][kt*32 + (l>>4)*16 + i][nt*16 + (l&15)]
// so a lane's whole WMMA B-fragment is 32 contiguous bytes.
__global__ __launch_bounds__(256) void k_convw(const float* __restrict__ src,
                                               unsigned short* __restrict__ dst,
                                               int K, int N) {
    const int e  = blockIdx.z;
    const int kt = blockIdx.y;           // K/32 tiles
    const int ng = blockIdx.x;           // groups of 16 n-tiles (256 cols)
    const int tid = threadIdx.x;
    const int NT = N >> 4;
    const int KT = K >> 5;

    __shared__ __align__(16) unsigned short sB[256 * 40];  // col-major, padded stride

    const size_t sbase = ((size_t)e * K + (size_t)kt * 32) * N + (size_t)ng * 256;
    for (int j = 0; j < 32; ++j)                       // coalesced fp32 reads
        sB[tid * 40 + j] = f2bf(src[sbase + (size_t)j * N + tid]);
    __syncthreads();

#pragma unroll
    for (int u = 0; u < 2; ++u) {
        int L = tid * 2 + u;                           // 0..511 lane-slots over 16 tiles
        int tile_local = L >> 5;
        int l = L & 31;
        const unsigned short* s = &sB[((tile_local * 16) + (l & 15)) * 40 + (l >> 4) * 16];
        size_t tile_global = ((size_t)e * NT + (size_t)(ng * 16 + tile_local)) * KT + kt;
        unsigned short* d = dst + tile_global * 512 + l * 16;
        *(uint4*)d       = *(const uint4*)s;
        *(uint4*)(d + 8) = *(const uint4*)(s + 8);
    }
}

// ---------------- kernel: routing (top-2 + softmax), one wave32 per token -------
__global__ __launch_bounds__(256) void k_routing(
    const float* __restrict__ x, const float* __restrict__ Wg,
    const float* __restrict__ bg, int* __restrict__ ridx, float* __restrict__ rw) {
    const int wave = threadIdx.x >> 5;
    const int lane = threadIdx.x & 31;
    const int tok  = blockIdx.x * 8 + wave;
    if (tok >= T_TOK) return;

    float acc[NE];
#pragma unroll
    for (int e = 0; e < NE; ++e) acc[e] = 0.f;
    const float* xr = x + (size_t)tok * DM;
    for (int d = lane; d < DM; d += 32) {
        float xv = xr[d];
        const float* wr = Wg + (size_t)d * NE;
#pragma unroll
        for (int e = 0; e < NE; ++e) acc[e] += xv * wr[e];
    }
#pragma unroll
    for (int off = 16; off >= 1; off >>= 1) {
#pragma unroll
        for (int e = 0; e < NE; ++e) acc[e] += __shfl_xor(acc[e], off, 32);
    }
    if (lane == 0) {
        float lg[NE];
#pragma unroll
        for (int e = 0; e < NE; ++e) lg[e] = acc[e] + bg[e];
        int i0 = 0; float v0 = lg[0];
#pragma unroll
        for (int e = 1; e < NE; ++e) if (lg[e] > v0) { v0 = lg[e]; i0 = e; }
        int i1 = (i0 == 0) ? 1 : 0; float v1 = -3.4e38f;
#pragma unroll
        for (int e = 0; e < NE; ++e) if (e != i0 && lg[e] > v1) { v1 = lg[e]; i1 = e; }
        float w0 = 1.f / (1.f + expf(v1 - v0));
        ridx[tok * 2 + 0] = i0; rw[tok * 2 + 0] = w0;
        ridx[tok * 2 + 1] = i1; rw[tok * 2 + 1] = 1.f - w0;
    }
}

// ---------------- kernel: build compacted per-expert pair lists (deterministic) --
__global__ __launch_bounds__(256) void k_build(
    const int* __restrict__ ridx, const float* __restrict__ rw,
    int* __restrict__ pair_tok, float* __restrict__ pair_w, int* __restrict__ offsets) {
    __shared__ int cnt[256][NE];
    __shared__ int totals[NE];
    __shared__ int base[NE + 1];
    const int tid = threadIdx.x;
    const int per = PAIRS / 256;

    int local[NE];
#pragma unroll
    for (int e = 0; e < NE; ++e) local[e] = 0;
    for (int i = 0; i < per; ++i) local[ridx[tid * per + i]]++;
#pragma unroll
    for (int e = 0; e < NE; ++e) cnt[tid][e] = local[e];
    __syncthreads();

    if (tid < NE) {
        int run = 0;
        for (int t = 0; t < 256; ++t) { int v = cnt[t][tid]; cnt[t][tid] = run; run += v; }
        totals[tid] = run;
    }
    __syncthreads();
    if (tid == 0) {
        int acc = 0;
        for (int e = 0; e < NE; ++e) { base[e] = acc; acc += (totals[e] + MTILE - 1) & ~(MTILE - 1); }
        base[NE] = acc;
        for (int e = 0; e <= NE; ++e) offsets[e] = base[e];
    }
    __syncthreads();

    int run2[NE];
#pragma unroll
    for (int e = 0; e < NE; ++e) run2[e] = 0;
    for (int i = 0; i < per; ++i) {
        int p = tid * per + i;
        int e = ridx[p];
        int pos = base[e] + cnt[tid][e] + run2[e]++;
        pair_tok[pos] = p >> 1;
        pair_w[pos]   = rw[p];
    }
    __syncthreads();
    for (int e = 0; e < NE; ++e) {
        for (int q = base[e] + totals[e] + tid; q < base[e + 1]; q += 256) {
            pair_tok[q] = 0;    // weight 0 -> contributes nothing
            pair_w[q]   = 0.f;
        }
    }
}

// ---------------- stage1: act = (xW0+b0) * silu(xW1+b1) -------------------------
// block: 32 pairs x 256 hidden cols; 8 waves x (2 m-tiles x 2 n-tiles x 2 matrices)
// No LDS, no barriers in the K loop: A/B fragments are direct global b128 loads.
__global__ __launch_bounds__(256) void k_stage1(
    const unsigned short* __restrict__ xb,
    const unsigned short* __restrict__ W0s, const unsigned short* __restrict__ W1s,
    const float* __restrict__ b0, const float* __restrict__ b1,
    const int* __restrict__ pair_tok, const int* __restrict__ offsets,
    unsigned short* __restrict__ act) {
    const int row0 = blockIdx.x * MTILE;
    if (row0 >= offsets[NE]) return;
    int e = 0;
    while (row0 >= offsets[e + 1]) ++e;          // padded offsets: multiples of 32

    const int tid = threadIdx.x, wave = tid >> 5, lane = tid & 31;
    const int lane15 = lane & 15, hi = lane >> 4;
    const int NT = DH >> 4;                      // 256 n-tiles
    const int KT = DM >> 5;                      // 32 k-tiles

    const int tokA = pair_tok[row0 + lane15];
    const int tokB = pair_tok[row0 + 16 + lane15];
    const unsigned short* a0p = xb + (size_t)tokA * DM + hi * 16;
    const unsigned short* a1p = xb + (size_t)tokB * DM + hi * 16;

    const int nt0 = blockIdx.y * 16 + wave * 2;
    const unsigned short* w00 = W0s + (((size_t)e * NT + nt0)     * KT) * 512 + lane * 16;
    const unsigned short* w01 = W0s + (((size_t)e * NT + nt0 + 1) * KT) * 512 + lane * 16;
    const unsigned short* w10 = W1s + (((size_t)e * NT + nt0)     * KT) * 512 + lane * 16;
    const unsigned short* w11 = W1s + (((size_t)e * NT + nt0 + 1) * KT) * 512 + lane * 16;

    const v8f vz = {0.f, 0.f, 0.f, 0.f, 0.f, 0.f, 0.f, 0.f};
    v8f ah[2][2] = {{vz, vz}, {vz, vz}};   // [mtile][ntile]
    v8f ag[2][2] = {{vz, vz}, {vz, vz}};

    for (int kk = 0; kk < DM; kk += 32) {
        v16bf A[2];
        A[0] = cat16(ldb8(a0p + kk), ldb8(a0p + kk + 8));
        A[1] = cat16(ldb8(a1p + kk), ldb8(a1p + kk + 8));
        v16bf Bh[2], Bg[2];
        Bh[0] = cat16(ldb8(w00), ldb8(w00 + 8));
        Bh[1] = cat16(ldb8(w01), ldb8(w01 + 8));
        Bg[0] = cat16(ldb8(w10), ldb8(w10 + 8));
        Bg[1] = cat16(ldb8(w11), ldb8(w11 + 8));
        __builtin_prefetch((const void*)(w00 + 2048), 0, 1);  // 4 tiles ahead
        __builtin_prefetch((const void*)(w01 + 2048), 0, 1);
        __builtin_prefetch((const void*)(w10 + 2048), 0, 1);
        __builtin_prefetch((const void*)(w11 + 2048), 0, 1);
#pragma unroll
        for (int mt = 0; mt < 2; ++mt) {
#pragma unroll
            for (int nt = 0; nt < 2; ++nt) {
                ah[mt][nt] = __builtin_amdgcn_wmma_f32_16x16x32_bf16(
                    false, A[mt], false, Bh[nt], (short)0, ah[mt][nt], false, false);
                ag[mt][nt] = __builtin_amdgcn_wmma_f32_16x16x32_bf16(
                    false, A[mt], false, Bg[nt], (short)0, ag[mt][nt], false, false);
            }
        }
        w00 += 512; w01 += 512; w10 += 512; w11 += 512;
    }

#pragma unroll
    for (int nt = 0; nt < 2; ++nt) {
        int col = blockIdx.y * 256 + wave * 32 + nt * 16 + lane15;
        float bh = b0[e * DH + col];
        float bs = b1[e * DH + col];
#pragma unroll
        for (int mt = 0; mt < 2; ++mt) {
#pragma unroll
            for (int r = 0; r < 8; ++r) {
                int m = mt * 16 + hi * 8 + r;
                float h = ah[mt][nt][r] + bh;
                float g = ag[mt][nt][r] + bs;
                act[(size_t)(row0 + m) * DH + col] = f2bf(h * (g / (1.f + expf(-g))));
            }
        }
    }
}

// ---------------- stage2: y = act @ W2 + b2 ; out[tok] += w*y -------------------
__global__ __launch_bounds__(256) void k_stage2(
    const unsigned short* __restrict__ act, const unsigned short* __restrict__ W2s,
    const float* __restrict__ b2,
    const int* __restrict__ pair_tok, const float* __restrict__ pair_w,
    const int* __restrict__ offsets, float* __restrict__ out) {
    const int row0 = blockIdx.x * MTILE;
    if (row0 >= offsets[NE]) return;
    int e = 0;
    while (row0 >= offsets[e + 1]) ++e;

    const int tid = threadIdx.x, wave = tid >> 5, lane = tid & 31;
    const int lane15 = lane & 15, hi = lane >> 4;
    const int NT = DM >> 4;                      // 64 n-tiles
    const int KT = DH >> 5;                      // 128 k-tiles

    __shared__ int   s_tok[MTILE];
    __shared__ float s_w[MTILE];
    if (tid < MTILE) { s_tok[tid] = pair_tok[row0 + tid]; s_w[tid] = pair_w[row0 + tid]; }
    __syncthreads();

    const unsigned short* a0p = act + (size_t)(row0 + lane15) * DH + hi * 16;
    const unsigned short* a1p = act + (size_t)(row0 + 16 + lane15) * DH + hi * 16;

    const int nt0 = blockIdx.y * 16 + wave * 2;
    const unsigned short* w0 = W2s + (((size_t)e * NT + nt0)     * KT) * 512 + lane * 16;
    const unsigned short* w1 = W2s + (((size_t)e * NT + nt0 + 1) * KT) * 512 + lane * 16;

    const v8f vz = {0.f, 0.f, 0.f, 0.f, 0.f, 0.f, 0.f, 0.f};
    v8f acc[2][2] = {{vz, vz}, {vz, vz}};        // [mtile][ntile]

    for (int kk = 0; kk < DH; kk += 32) {
        v16bf A[2];
        A[0] = cat16(ldb8(a0p + kk), ldb8(a0p + kk + 8));
        A[1] = cat16(ldb8(a1p + kk), ldb8(a1p + kk + 8));
        v16bf B[2];
        B[0] = cat16(ldb8(w0), ldb8(w0 + 8));
        B[1] = cat16(ldb8(w1), ldb8(w1 + 8));
        __builtin_prefetch((const void*)(w0 + 2048), 0, 1);
        __builtin_prefetch((const void*)(w1 + 2048), 0, 1);
#pragma unroll
        for (int mt = 0; mt < 2; ++mt) {
#pragma unroll
            for (int nt = 0; nt < 2; ++nt) {
                acc[mt][nt] = __builtin_amdgcn_wmma_f32_16x16x32_bf16(
                    false, A[mt], false, B[nt], (short)0, acc[mt][nt], false, false);
            }
        }
        w0 += 512; w1 += 512;
    }

#pragma unroll
    for (int nt = 0; nt < 2; ++nt) {
        int col = blockIdx.y * 256 + wave * 32 + nt * 16 + lane15;
        float bb = b2[e * DM + col];
#pragma unroll
        for (int mt = 0; mt < 2; ++mt) {
#pragma unroll
            for (int r = 0; r < 8; ++r) {
                int m = mt * 16 + hi * 8 + r;
                float val = s_w[m] * (acc[mt][nt][r] + bb);
                atomicAdd(&out[(size_t)s_tok[m] * DM + col], val);  // 2 adds/elem: order-safe
            }
        }
    }
}

// ---------------- launch ----------------
extern "C" void kernel_launch(void* const* d_in, const int* in_sizes, int n_in,
                              void* d_out, int out_size, void* d_ws, size_t ws_size,
                              hipStream_t stream) {
    const float* x  = (const float*)d_in[0];
    const float* Wg = (const float*)d_in[1];
    const float* bg = (const float*)d_in[2];
    const float* W0 = (const float*)d_in[3];
    const float* b0 = (const float*)d_in[4];
    const float* W1 = (const float*)d_in[5];
    const float* b1 = (const float*)d_in[6];
    const float* W2 = (const float*)d_in[7];
    const float* b2 = (const float*)d_in[8];
    float* out = (float*)d_out;

    char* ws = (char*)d_ws;
    size_t o = 0;
    int*   offsets  = (int*)(ws + o); o += 64;
    int*   ridx     = (int*)(ws + o); o += (size_t)PAIRS * 4;
    float* rw       = (float*)(ws + o); o += (size_t)PAIRS * 4;
    int*   pair_tok = (int*)(ws + o); o += (size_t)MAXP * 4;
    float* pair_w   = (float*)(ws + o); o += (size_t)MAXP * 4;
    o = (o + 255) & ~(size_t)255;
    unsigned short* xb  = (unsigned short*)(ws + o); o += (size_t)T_TOK * DM * 2;
    unsigned short* W0s = (unsigned short*)(ws + o); o += (size_t)NE * DM * DH * 2;
    unsigned short* W1s = (unsigned short*)(ws + o); o += (size_t)NE * DM * DH * 2;
    unsigned short* W2s = (unsigned short*)(ws + o); o += (size_t)NE * DH * DM * 2;
    unsigned short* act = (unsigned short*)(ws + o); o += (size_t)MAXP * DH * 2;

    k_zero<<<(T_TOK * DM + 255) / 256, 256, 0, stream>>>(out, T_TOK * DM);
    k_convx<<<(T_TOK * DM / 4 + 255) / 256, 256, 0, stream>>>(x, xb, T_TOK * DM);
    k_convw<<<dim3(DH / 256, DM / 32, NE), 256, 0, stream>>>(W0, W0s, DM, DH);
    k_convw<<<dim3(DH / 256, DM / 32, NE), 256, 0, stream>>>(W1, W1s, DM, DH);
    k_convw<<<dim3(DM / 256, DH / 32, NE), 256, 0, stream>>>(W2, W2s, DH, DM);
    k_routing<<<T_TOK / 8, 256, 0, stream>>>(x, Wg, bg, ridx, rw);
    k_build<<<1, 256, 0, stream>>>(ridx, rw, pair_tok, pair_w, offsets);
    k_stage1<<<dim3(MAXTILES, DH / 256), 256, 0, stream>>>(xb, W0s, W1s, b0, b1,
                                                           pair_tok, offsets, act);
    k_stage2<<<dim3(MAXTILES, DM / 256), 256, 0, stream>>>(act, W2s, b2, pair_tok,
                                                           pair_w, offsets, out);
}
